// LinearReshape_38053410243068
// MI455X (gfx1250) — matrix operand order repaired and checked by reading, not compiled, not run on hardware
//
#include <hip/hip_runtime.h>
#include <stdint.h>

typedef __attribute__((ext_vector_type(16))) __bf16 v16bf;
typedef __attribute__((ext_vector_type(8)))  float  v8f;

__device__ __forceinline__ uint32_t f2bf(float x) {
  // round-to-nearest-even f32 -> bf16 (NaN edge cases ignored; inputs are normal)
  uint32_t u = __float_as_uint(x);
  u += 0x7FFFu + ((u >> 16) & 1u);
  return u >> 16;
}

// ---------------- W f32 -> bf16 prep (runs once per launch, tiny) ----------
__global__ void wconv_kernel(const float* __restrict__ W,
                             uint16_t* __restrict__ Wb, int n) {
  int i = blockIdx.x * blockDim.x + threadIdx.x;
  if (i < n) Wb[i] = (uint16_t)f2bf(W[i]);
}

// ---------------- fused LayerNorm + GEMM(WMMA bf16) + SiLU + unpatchify ----
#define LDS_STRIDE 792   // 768 + 24 pad: row stride 1584B = 396 dwords; 396%64=12
                         // -> lanes 0..15 (Δm*12 mod 64 != 0) conflict-free A loads

__global__ __launch_bounds__(256)
void fused_ln_gemm_silu_kernel(const float* __restrict__ x,
                               const float* __restrict__ lnw,
                               const float* __restrict__ lnb,
                               const uint16_t* __restrict__ Wb,
                               const float* __restrict__ bias,
                               float* __restrict__ out) {
  __shared__ uint16_t xn_s[32 * LDS_STRIDE];   // 50688 B normalized bf16 tile

  const int bi   = blockIdx.x >> 1;   // batch index
  const int hi   = blockIdx.x & 1;    // which 32-row half of the 64-row tile
  const int lane = threadIdx.x & 31;
  const int wave = threadIdx.x >> 5;  // 8 waves

  // ---------- Phase 1: LayerNorm 4 rows per wave, bf16 into LDS ----------
  for (int i = 0; i < 4; ++i) {
    const int r = wave * 4 + i;                       // local row 0..31
    const float4* xr =
        (const float4*)(x + (size_t)(bi * 64 + hi * 32 + r) * 768);
    float4 xv[6];
    float s = 0.f, ss = 0.f;
#pragma unroll
    for (int j = 0; j < 6; ++j) {
      xv[j] = xr[lane + 32 * j];                      // coalesced 16B loads
      s  += xv[j].x + xv[j].y + xv[j].z + xv[j].w;
      ss += xv[j].x * xv[j].x + xv[j].y * xv[j].y +
            xv[j].z * xv[j].z + xv[j].w * xv[j].w;
    }
#pragma unroll
    for (int m = 16; m > 0; m >>= 1) {                // wave32 reduction
      s  += __shfl_xor(s,  m, 32);
      ss += __shfl_xor(ss, m, 32);
    }
    const float mu   = s * (1.0f / 768.0f);
    const float var  = ss * (1.0f / 768.0f) - mu * mu;
    const float rstd = rsqrtf(var + 1e-5f);
#pragma unroll
    for (int j = 0; j < 6; ++j) {
      const int vi = lane + 32 * j;                   // float4 index in row
      const float4 wv = ((const float4*)lnw)[vi];
      const float4 bv = ((const float4*)lnb)[vi];
      float4 nv;
      nv.x = (xv[j].x - mu) * rstd * wv.x + bv.x;
      nv.y = (xv[j].y - mu) * rstd * wv.y + bv.y;
      nv.z = (xv[j].z - mu) * rstd * wv.z + bv.z;
      nv.w = (xv[j].w - mu) * rstd * wv.w + bv.w;
      uint2 p;
      p.x = f2bf(nv.x) | (f2bf(nv.y) << 16);
      p.y = f2bf(nv.z) | (f2bf(nv.w) << 16);
      *(uint2*)&xn_s[r * LDS_STRIDE + 4 * vi] = p;    // 8B aligned LDS store
    }
  }
  __syncthreads();

  // ---------- Phase 2: one 16x16 output tile per wave ----------
  const int mi    = wave >> 2;                 // 0..1  (M tile)
  const int ni    = wave & 3;                  // 0..3  (N tile)
  const int l16   = lane & 15;
  const int hiH   = (lane < 16) ? 0 : 1;
  const int mrow  = mi * 16 + l16;             // A-fragment row (both halves)
  const int ncol  = ni * 16 + l16;             // B-fragment column / output ch
  const int aOff  = hiH * 8;                   // A: lanes>=16 start at K=8
  const int bOff  = hiH * 8;                   // B: lanes>=16 start at K=8

  union Frag { v16bf v; uint4 u[2]; };
  Frag a, b;
  v8f acc = {};

  const uint16_t* __restrict__ arow = &xn_s[mrow * LDS_STRIDE];
  const uint16_t* __restrict__ wrow = Wb + ncol * 768;

#pragma unroll 4
  for (int kk = 0; kk < 24; ++kk) {
    const int k0 = kk * 32;
    // A (16x32 bf16): elems 0..7 -> K = k0+aOff+0..7, elems 8..15 -> +16
    a.u[0] = *(const uint4*)&arow[k0 + aOff];
    a.u[1] = *(const uint4*)&arow[k0 + 16 + aOff];
    // B (32x16 bf16): elems 0..7 -> K = k0+bOff+0..7, elems 8..15 -> +16
    b.u[0] = *(const uint4*)&wrow[k0 + bOff];
    b.u[1] = *(const uint4*)&wrow[k0 + bOff + 16];
    acc = __builtin_amdgcn_wmma_f32_16x16x32_bf16(
        /*neg_a=*/false, a.v, /*neg_b=*/false, b.v,
        /*c_mod=*/(short)0, acc, /*reuse_a=*/false, /*reuse_b=*/false);
  }

  // ---------- epilogue: bias + SiLU + unpatchify store ----------
  const float bn = bias[ncol];
  const int d = ncol >> 4;
  const int e = (ncol >> 2) & 3;
  const int f = ncol & 3;
  float* __restrict__ outb = out + (size_t)bi * 4096 + d * 1024 + e * 32 + f;
#pragma unroll
  for (int v = 0; v < 8; ++v) {
    const int c = hi * 32 + mi * 16 + v + hiH * 8;    // patch index 0..63
    const int p = c >> 3, q = c & 7;
    float y = acc[v] + bn;
    y = y / (1.0f + __expf(-y));                      // SiLU
    outb[p * 128 + q * 4] = y;   // (p*4)*32 + q*4 within [4,32,32]
  }
}

extern "C" void kernel_launch(void* const* d_in, const int* in_sizes, int n_in,
                              void* d_out, int out_size, void* d_ws, size_t ws_size,
                              hipStream_t stream) {
  const float* x   = (const float*)d_in[0];   // [B,64,768]
  const float* lnw = (const float*)d_in[1];   // [768]
  const float* lnb = (const float*)d_in[2];   // [768]
  const float* W   = (const float*)d_in[3];   // [64,768]
  const float* b   = (const float*)d_in[4];   // [64]
  float* out = (float*)d_out;                 // [B,4,32,32]
  uint16_t* Wb = (uint16_t*)d_ws;             // 64*768 bf16 = 96 KB scratch

  const int B = in_sizes[0] / (64 * 768);
  const int wn = 64 * 768;

  wconv_kernel<<<(wn + 255) / 256, 256, 0, stream>>>(W, Wb, wn);
  fused_ln_gemm_silu_kernel<<<dim3(B * 2), dim3(256), 0, stream>>>(
      x, lnw, lnb, Wb, b, out);
}
